// SelfAttention_48455821034015
// MI455X (gfx1250) — compile-verified, matrix operand order checked by reading, not run on hardware
//
#include <hip/hip_runtime.h>
#include <hip/hip_bf16.h>
#include <cstdint>

typedef __attribute__((ext_vector_type(16))) _Float16 v16h;
typedef __attribute__((ext_vector_type(8)))  _Float16 v8h;
typedef __attribute__((ext_vector_type(8)))  float    v8f;
typedef __attribute__((ext_vector_type(4)))  float    v4f;
typedef __attribute__((ext_vector_type(2)))  float    v2f;

#define BN_EPS 1e-5f
#define B_    4
#define C_    256
#define CQK_  32
#define N_    4096

static __device__ __forceinline__ v16h make16(v8h lo, v8h hi) {
  return __builtin_shufflevector(lo, hi, 0,1,2,3,4,5,6,7,8,9,10,11,12,13,14,15);
}

// Load one 16x32 f16 WMMA operand tile in the CDNA5 per-lane register layout.
// Memory is "row-major" with leading dimension ld: A tiles pass rows=M, B tiles
// pass rows=N-columns (i.e. column-major K x 16 data). Per lane: row = lane%16,
// K chunks [8*hi, 8*hi+8) and [16+8*hi, 16+8*hi+8), each a contiguous b128.
static __device__ __forceinline__ v16h ld_tile16(const _Float16* base, int ld) {
  const int lane = threadIdx.x & 31;
  const int row  = lane & 15;
  const int koff = (lane < 16) ? 0 : 8;
  const _Float16* p = base + row * ld + koff;
  v8h lo = *(const v8h*)(p);
  v8h hi = *(const v8h*)(p + 16);
  return make16(lo, hi);
}

// ---------------------------------------------------------------------------
// Kernel 1: fused q/k/v 1x1-conv + BatchNorm, f32 in -> f16 out (WMMA-ready).
// Grid: B * (N/64) blocks of 128 threads (4 waves, one 16-row n-subtile each).
// ---------------------------------------------------------------------------
__global__ __launch_bounds__(128) void proj_kernel(
    const float* __restrict__ x,
    const float* __restrict__ Wq, const float* __restrict__ Wk,
    const float* __restrict__ Wv,
    const float* __restrict__ bnq_w, const float* __restrict__ bnq_b,
    const float* __restrict__ bnq_m, const float* __restrict__ bnq_v,
    const float* __restrict__ bnk_w, const float* __restrict__ bnk_b,
    const float* __restrict__ bnk_m, const float* __restrict__ bnk_v,
    const float* __restrict__ bnv_w, const float* __restrict__ bnv_b,
    const float* __restrict__ bnv_m, const float* __restrict__ bnv_v,
    _Float16* __restrict__ qh, _Float16* __restrict__ kh,
    _Float16* __restrict__ vh)
{
  const int b     = blockIdx.x / (N_ / 64);
  const int ntile = blockIdx.x % (N_ / 64);
  const int wave  = threadIdx.x >> 5;
  const int lane  = threadIdx.x & 31;
  const int nbase = ntile * 64 + wave * 16;
  const float* xb = x + (size_t)b * C_ * N_;

  // A = x^T tile [16 n x 32 c] per K-chunk, gathered (stride-N) and cast f16.
  const int nA = nbase + (lane & 15);
  const int cA = (lane < 16) ? 0 : 8;
  v16h A[8];
#pragma unroll
  for (int kc = 0; kc < 8; ++kc) {
    v16h a;
#pragma unroll
    for (int i = 0; i < 8; ++i) {
      a[i]     = (_Float16)xb[(size_t)(kc * 32 + cA + i)      * N_ + nA];
      a[i + 8] = (_Float16)xb[(size_t)(kc * 32 + 16 + cA + i) * N_ + nA];
    }
    A[kc] = a;
  }

  // 20 output 16-col tiles: q(2) + k(2) + v(16), each accumulated over K=256.
#pragma unroll 1
  for (int t = 0; t < 20; ++t) {
    const float *W, *sw, *sb, *sm, *sv;
    int dbase;
    if (t < 2)      { W = Wq; dbase = t * 16;
                      sw = bnq_w; sb = bnq_b; sm = bnq_m; sv = bnq_v; }
    else if (t < 4) { W = Wk; dbase = (t - 2) * 16;
                      sw = bnk_w; sb = bnk_b; sm = bnk_m; sv = bnk_v; }
    else            { W = Wv; dbase = (t - 4) * 16;
                      sw = bnv_w; sb = bnv_b; sm = bnv_m; sv = bnv_v; }
    const int d = dbase + (lane & 15);

    v8f acc = {};
#pragma unroll
    for (int kc = 0; kc < 8; ++kc) {
      // B tile: W rows are output channels (B columns), K=c contiguous.
      const float* p = W + (size_t)d * C_ + kc * 32 + cA;
      v4f b0 = *(const v4f*)(p);
      v4f b1 = *(const v4f*)(p + 4);
      v4f b2 = *(const v4f*)(p + 16);
      v4f b3 = *(const v4f*)(p + 20);
      v16h bb;
#pragma unroll
      for (int i = 0; i < 4; ++i) {
        bb[i]      = (_Float16)b0[i];
        bb[i + 4]  = (_Float16)b1[i];
        bb[i + 8]  = (_Float16)b2[i];
        bb[i + 12] = (_Float16)b3[i];
      }
      acc = __builtin_amdgcn_wmma_f32_16x16x32_f16(
          false, A[kc], false, bb, (short)0, acc, false, false);
    }

    // BatchNorm epilogue for this lane's output channel.
    const float scale = sw[d] * rsqrtf(sv[d] + BN_EPS);
    const float bias  = sb[d] - sm[d] * scale;
    const int   nrow0 = nbase + ((lane < 16) ? 0 : 8);
    if (t < 4) {
      _Float16* outp = (t < 2 ? qh : kh) + (size_t)b * N_ * CQK_;
#pragma unroll
      for (int r = 0; r < 8; ++r) {
        float y = acc[r] * scale + bias;
        outp[(size_t)(nrow0 + r) * CQK_ + d] = (_Float16)y;
      }
    } else {
      _Float16* outp = vh + ((size_t)b * C_ + d) * N_;
#pragma unroll
      for (int r = 0; r < 8; ++r) {
        float y = acc[r] * scale + bias;
        outp[nrow0 + r] = (_Float16)y;
      }
    }
  }
}

// ---------------------------------------------------------------------------
// Kernel 2: per-row softmax statistics (running max + sum of exp), online.
// Grid: B * (N/128) blocks of 256 threads (8 waves, one 16-row m-tile each).
// ---------------------------------------------------------------------------
__global__ __launch_bounds__(256) void stats_kernel(
    const _Float16* __restrict__ qh, const _Float16* __restrict__ kh,
    float* __restrict__ stats)
{
  const int b     = blockIdx.x / (N_ / 128);
  const int mt    = blockIdx.x % (N_ / 128);
  const int wave  = threadIdx.x >> 5;
  const int lane  = threadIdx.x & 31;
  const int mbase = mt * 128 + wave * 16;

  const v16h Aq = ld_tile16(qh + ((size_t)b * N_ + mbase) * CQK_, CQK_);

  float M[8], L[8];
#pragma unroll
  for (int r = 0; r < 8; ++r) { M[r] = -1e30f; L[r] = 0.f; }

  for (int nt = 0; nt < N_ / 16; ++nt) {
    v16h Bk = ld_tile16(kh + ((size_t)b * N_ + nt * 16) * CQK_, CQK_);
    v8f S = {};
    S = __builtin_amdgcn_wmma_f32_16x16x32_f16(
        false, Aq, false, Bk, (short)0, S, false, false);
#pragma unroll
    for (int r = 0; r < 8; ++r) {
      float s  = S[r];
      float tm = s;
      tm = fmaxf(tm, __shfl_xor(tm, 1, 16));
      tm = fmaxf(tm, __shfl_xor(tm, 2, 16));
      tm = fmaxf(tm, __shfl_xor(tm, 4, 16));
      tm = fmaxf(tm, __shfl_xor(tm, 8, 16));
      float nm = fmaxf(M[r], tm);
      float p  = __expf(s - nm);
      p += __shfl_xor(p, 1, 16);
      p += __shfl_xor(p, 2, 16);
      p += __shfl_xor(p, 4, 16);
      p += __shfl_xor(p, 8, 16);
      L[r] = L[r] * __expf(M[r] - nm) + p;
      M[r] = nm;
    }
  }

  if ((lane & 15) == 0) {
    const int hi = lane >> 4;
#pragma unroll
    for (int r = 0; r < 8; ++r) {
      const int m = mbase + r + 8 * hi;
      stats[((size_t)b * N_ + m) * 2 + 0] = M[r];
      stats[((size_t)b * N_ + m) * 2 + 1] = L[r];
    }
  }
}

// ---------------------------------------------------------------------------
// Kernel 3: streaming P = exp(S - max) and out = gamma * (P @ v^T)/L + x.
// Grid: B * (N/32) blocks of 256 threads.
// Wave w: m-subtile = w&1 (16 rows), c-quarter = w>>1 (64 channels, 4 tiles).
// Waves 0/1 produce P (S WMMA + exp) into double-buffered LDS; all 8 waves
// consume it as the A operand of the PV WMMA.
// ---------------------------------------------------------------------------
__global__ __launch_bounds__(256) void attn_out_kernel(
    const _Float16* __restrict__ qh, const _Float16* __restrict__ kh,
    const _Float16* __restrict__ vh, const float* __restrict__ stats,
    const float* __restrict__ x, const float* __restrict__ gamma,
    float* __restrict__ out)
{
  __shared__ __align__(16) _Float16 P[2][2][16 * 32]; // [buf][msub][m][n]

  const int b    = blockIdx.x / (N_ / 32);
  const int mt   = blockIdx.x % (N_ / 32);
  const int wave = threadIdx.x >> 5;
  const int lane = threadIdx.x & 31;
  const int msub = wave & 1;
  const int cq   = wave >> 1;               // 0..3
  const int mbase = mt * 32 + msub * 16;
  const int hi    = (lane < 16) ? 0 : 1;

  // Row softmax stats for this lane's 8 rows (m = mbase + r + 8*hi).
  float Mv[8], Lv[8];
#pragma unroll
  for (int r = 0; r < 8; ++r) {
    const int m = mbase + r + 8 * hi;
    v2f st = *(const v2f*)(stats + ((size_t)b * N_ + m) * 2);
    Mv[r] = st[0];
    Lv[r] = st[1];
  }

  v16h Aq = {};
  if (wave < 2)
    Aq = ld_tile16(qh + ((size_t)b * N_ + mbase) * CQK_, CQK_);

  v8f acc[4] = {};

  for (int nc = 0; nc < N_ / 32; ++nc) {
    const int buf = nc & 1;
    if (wave < 2) {  // wave-uniform branch: EXEC stays all-ones for WMMA
#pragma unroll
      for (int sub = 0; sub < 2; ++sub) {
        v16h Bk = ld_tile16(
            kh + ((size_t)b * N_ + nc * 32 + sub * 16) * CQK_, CQK_);
        v8f S = {};
        S = __builtin_amdgcn_wmma_f32_16x16x32_f16(
            false, Aq, false, Bk, (short)0, S, false, false);
        const int ncol = (lane & 15) + sub * 16;
#pragma unroll
        for (int r = 0; r < 8; ++r) {
          float p = __expf(S[r] - Mv[r]);
          P[buf][msub][(r + 8 * hi) * 32 + ncol] = (_Float16)p;
        }
      }
    }
    __syncthreads();

    // PV: A = P tile (16m x 32n) from LDS, B = v^T (32n x 16c) from global.
    {
      const int row  = lane & 15;
      const int koff = hi ? 8 : 0;
      const _Float16* pp = &P[buf][msub][row * 32 + koff];
      v8h plo = *(const v8h*)(pp);
      v8h phi = *(const v8h*)(pp + 16);
      v16h Ap = make16(plo, phi);
#pragma unroll
      for (int ct = 0; ct < 4; ++ct) {
        const int c = cq * 64 + ct * 16 + (lane & 15);
        const _Float16* vb = vh + ((size_t)b * C_ + c) * N_ + nc * 32 + koff;
        v8h vlo = *(const v8h*)(vb);
        v8h vhi = *(const v8h*)(vb + 16);
        v16h Bv = make16(vlo, vhi);
        acc[ct] = __builtin_amdgcn_wmma_f32_16x16x32_f16(
            false, Ap, false, Bv, (short)0, acc[ct], false, false);
      }
    }
  }

  // Epilogue: out = gamma * acc / L + x  (D rows = m, cols = c).
  const float g = gamma[0];
#pragma unroll
  for (int ct = 0; ct < 4; ++ct) {
    const int c = cq * 64 + ct * 16 + (lane & 15);
    const float* xr  = x   + ((size_t)b * C_ + c) * N_ + mbase + 8 * hi;
    float*       orw = out + ((size_t)b * C_ + c) * N_ + mbase + 8 * hi;
#pragma unroll
    for (int r = 0; r < 8; ++r) {
      float y = acc[ct][r] / Lv[r];
      orw[r] = g * y + xr[r];
    }
  }
}

// ---------------------------------------------------------------------------
extern "C" void kernel_launch(void* const* d_in, const int* in_sizes, int n_in,
                              void* d_out, int out_size, void* d_ws, size_t ws_size,
                              hipStream_t stream) {
  (void)in_sizes; (void)n_in; (void)out_size; (void)ws_size;
  const float* x     = (const float*)d_in[0];
  const float* Wq    = (const float*)d_in[1];
  const float* Wk    = (const float*)d_in[2];
  const float* Wv    = (const float*)d_in[3];
  const float* bnq_w = (const float*)d_in[4];
  const float* bnq_b = (const float*)d_in[5];
  const float* bnq_m = (const float*)d_in[6];
  const float* bnq_v = (const float*)d_in[7];
  const float* bnk_w = (const float*)d_in[8];
  const float* bnk_b = (const float*)d_in[9];
  const float* bnk_m = (const float*)d_in[10];
  const float* bnk_v = (const float*)d_in[11];
  const float* bnv_w = (const float*)d_in[12];
  const float* bnv_b = (const float*)d_in[13];
  const float* bnv_m = (const float*)d_in[14];
  const float* bnv_v = (const float*)d_in[15];
  const float* gamma = (const float*)d_in[16];
  float* out = (float*)d_out;

  // Workspace layout (f16 q/k/v + f32 row stats): ~10.2 MB total.
  char* ws = (char*)d_ws;
  _Float16* qh    = (_Float16*)(ws);                           // 1 MB
  _Float16* kh    = (_Float16*)(ws + ((size_t)1 << 20));       // 1 MB
  _Float16* vh    = (_Float16*)(ws + ((size_t)2 << 20));       // 8 MB
  float*    stats = (float*)   (ws + ((size_t)10 << 20));      // 128 KB

  proj_kernel<<<B_ * (N_ / 64), 128, 0, stream>>>(
      x, Wq, Wk, Wv,
      bnq_w, bnq_b, bnq_m, bnq_v,
      bnk_w, bnk_b, bnk_m, bnk_v,
      bnv_w, bnv_b, bnv_m, bnv_v,
      qh, kh, vh);

  stats_kernel<<<B_ * (N_ / 128), 256, 0, stream>>>(qh, kh, stats);

  attn_out_kernel<<<B_ * (N_ / 32), 256, 0, stream>>>(
      qh, kh, vh, stats, x, gamma, out);
}
